// MultiHeadAttention_21131239096906
// MI455X (gfx1250) — matrix operand order repaired and checked by reading, not compiled, not run on hardware
//
#include <hip/hip_runtime.h>
#include <hip/hip_bf16.h>

// ---------------------------------------------------------------------------
// MHA forward for gfx1250 (MI455X), wave32.
// Matrix ops: v_wmma_f32_16x16x32_bf16.  Tile staging in the attention kernel
// uses global_load_async_to_lds_b128 (ASYNCcnt) -- the CDNA5 async path.
// B=4, S=2048, E=512, H=8, D=64, SCALE=181.0 (faithful to reference bug).
// ---------------------------------------------------------------------------

typedef __attribute__((ext_vector_type(16))) __bf16 v16bf;
typedef __attribute__((ext_vector_type(8)))  float  v8f;

#define WMMA_BF16(a, b, c) \
    __builtin_amdgcn_wmma_f32_16x16x32_bf16(false, (a), false, (b), (short)0, (c), false, false)

namespace mha {
constexpr int Bn = 4;
constexpr int Sn = 2048;
constexpr int En = 512;
constexpr int Hn = 8;
constexpr int Dn = 64;
constexpr float INV_SCALE = 1.0f / 181.0f;   // E // (H**0.5) == 181
constexpr float LN_EPS = 1e-5f;
constexpr int PADP = 40;   // proj LDS row pad (80 B, 16B multiple)
constexpr int PADA = 72;   // attn LDS row pad (144 B, 16B multiple)
}

// A-fragment K index per CDNA5 ISA 16-bit A 16x32 layout
// (lanes 0-15: K0..7,16..23 ; lanes 16-31: K8..15,24..31)
__device__ __forceinline__ int a_kidx(int e, bool hi) {
    return (e < 8) ? ((hi ? 8 : 0) + e) : ((hi ? 24 : 16) + (e - 8));
}
// B-fragment K index (32x16): same order as A (lanes 0-15: K0..7,16..23 ; lanes 16-31: K8..15,24..31)
__device__ __forceinline__ int b_kidx(int e, bool hi) {
    return (e < 8) ? ((hi ? 8 : 0) + e) : ((hi ? 24 : 16) + (e - 8));
}

// fp32x4 -> 4 contiguous bf16 in LDS (lowers to packed cvt + ds_store_b64)
__device__ __forceinline__ void cvt_store4(__bf16* dst, const float4 v) {
    dst[0] = (__bf16)v.x; dst[1] = (__bf16)v.y;
    dst[2] = (__bf16)v.z; dst[3] = (__bf16)v.w;
}

// CDNA5 async global->LDS 16-byte copy (per-lane), tracked by ASYNCcnt.
__device__ __forceinline__ void async_b128(unsigned lds_off, const void* gaddr) {
    asm volatile("global_load_async_to_lds_b128 %0, %1, off"
                 :: "v"(lds_off), "v"((unsigned long long)(uintptr_t)gaddr)
                 : "memory");
}
__device__ __forceinline__ void wait_async0() {
    asm volatile("s_wait_asynccnt 0x0" ::: "memory");
}

// ---------------------------------------------------------------------------
// Kernel 1: Y = X @ W^T + bias, row-masked, per-head LayerNorm over D=64,
// emit bf16 [B,H,S,D].  Tile: 32 rows x 64 cols (one head). 8 waves.
// ---------------------------------------------------------------------------
__global__ __launch_bounds__(256)
void proj_ln_kernel(const float* __restrict__ X, const float* __restrict__ W,
                    const float* __restrict__ bias, const unsigned char* __restrict__ rmask,
                    const float* __restrict__ gamma, const float* __restrict__ beta,
                    __hip_bfloat16* __restrict__ Out) {
    using namespace mha;
    __shared__ __align__(16) __bf16 As[32][PADP];
    __shared__ __align__(16) __bf16 Bs[64][PADP];
    __shared__ float Cs[32][65];

    const int tid  = threadIdx.x;
    const int wave = tid >> 5;
    const int lane = tid & 31;
    const bool hi  = lane >= 16;
    const int tm = wave >> 2;          // 0..1  (16-row tile)
    const int tn = wave & 3;           // 0..3  (16-col tile)
    const int m0 = blockIdx.x * 32;    // global row base (B*S rows)
    const int h  = blockIdx.y;         // head
    const int n0 = h * Dn;             // global col base

    v8f acc = {};
    for (int kk = 0; kk < En; kk += 32) {
        {   // A: 32x32 fp32 -> bf16; 256 float4 chunks, one per thread
            const int r = tid >> 3, c4 = (tid & 7) * 4;
            const float4 xv = *(const float4*)&X[(size_t)(m0 + r) * En + kk + c4];
            cvt_store4(&As[r][c4], xv);
        }
#pragma unroll
        for (int i = tid; i < 512; i += 256) {  // B: 64x32 fp32 -> bf16
            const int r = i >> 3, c4 = (i & 7) * 4;
            const float4 wv = *(const float4*)&W[(size_t)(n0 + r) * En + kk + c4];
            cvt_store4(&Bs[r][c4], wv);
        }
        __syncthreads();
        v16bf a, b;
        const int mA = tm * 16 + (lane & 15);
        const int nB = tn * 16 + (lane & 15);
#pragma unroll
        for (int e = 0; e < 16; ++e) {
            a[e] = As[mA][a_kidx(e, hi)];
            b[e] = Bs[nB][b_kidx(e, hi)];
        }
        acc = WMMA_BF16(a, b, acc);
        __syncthreads();
    }

    // epilogue: bias + row mask -> LDS fp32
    {
        const int n_loc = tn * 16 + (lane & 15);
        const float bv = bias[n0 + n_loc];
#pragma unroll
        for (int r = 0; r < 8; ++r) {
            const int m_loc = tm * 16 + r + (hi ? 8 : 0);
            const float mk = rmask[m0 + m_loc] ? 1.0f : 0.0f;
            Cs[m_loc][n_loc] = (acc[r] + bv) * mk;
        }
    }
    __syncthreads();

    // per-head LayerNorm over the 64-wide row, then store bf16 [B,H,S,D]
    if (tid < 32) {
        const int m_loc = tid;
        float mu = 0.0f;
#pragma unroll
        for (int c = 0; c < Dn; ++c) mu += Cs[m_loc][c];
        mu *= (1.0f / Dn);
        float var = 0.0f;
#pragma unroll
        for (int c = 0; c < Dn; ++c) {
            float d = Cs[m_loc][c] - mu;
            var += d * d;
        }
        var *= (1.0f / Dn);
        const float rs = rsqrtf(var + LN_EPS);
        const int gm = m0 + m_loc;
        const int bb = gm / Sn, ss = gm % Sn;
        const size_t obase = (((size_t)bb * Hn + h) * Sn + ss) * Dn;
#pragma unroll
        for (int c = 0; c < Dn; ++c) {
            float y = (Cs[m_loc][c] - mu) * rs * gamma[c] + beta[c];
            Out[obase + c] = __float2bfloat16(y);
        }
    }
}

// ---------------------------------------------------------------------------
// Kernel 2: flash attention per (b,h).  64 Q-rows per block, 4 waves.
// Tile staging via global_load_async_to_lds_b128 (ASYNCcnt).
// Online softmax over 32 key tiles of 64.  Context -> bf16 [B,S,H,D].
// ---------------------------------------------------------------------------
__global__ __launch_bounds__(128)
void attn_kernel(const __hip_bfloat16* __restrict__ Qg, const __hip_bfloat16* __restrict__ Kg,
                 const __hip_bfloat16* __restrict__ Vg,
                 const unsigned char* __restrict__ qmask, const unsigned char* __restrict__ kmask,
                 __hip_bfloat16* __restrict__ Ctx) {
    using namespace mha;
    __shared__ __align__(16) __bf16 Qs[64][PADA];
    __shared__ __align__(16) __bf16 Ks[64][PADA];
    __shared__ __align__(16) __bf16 Vs[64][PADA];
    __shared__ __align__(16) __bf16 Ps[4][16][PADA];
    __shared__ float kmb[64];

    const int tid  = threadIdx.x;
    const int wave = tid >> 5;
    const int lane = tid & 31;
    const bool hi  = lane >= 16;
    const int bh = blockIdx.y;                 // b*H + h
    const int bb = bh / Hn, h = bh % Hn;
    const int q0 = blockIdx.x * 64;
    const size_t base = (size_t)bh * Sn * Dn;
    const __bf16* Qp = reinterpret_cast<const __bf16*>(Qg) + base;
    const __bf16* Kp = reinterpret_cast<const __bf16*>(Kg) + base;
    const __bf16* Vp = reinterpret_cast<const __bf16*>(Vg) + base;

    const unsigned qsOff = (unsigned)(uintptr_t)&Qs[0][0];
    const unsigned ksOff = (unsigned)(uintptr_t)&Ks[0][0];
    const unsigned vsOff = (unsigned)(uintptr_t)&Vs[0][0];

    // Q tile: 64 rows x 64 bf16 = 512 x 16B chunks
#pragma unroll
    for (int i = tid; i < 512; i += 128) {
        const int r = i >> 3, c8 = (i & 7) * 8;
        async_b128(qsOff + (unsigned)((r * PADA + c8) * 2),
                   Qp + (size_t)(q0 + r) * Dn + c8);
    }
    wait_async0();
    __syncthreads();

    // Q strip fragments (16 rows per wave, D=64 -> two K32 chunks), resident.
    v16bf qf[2];
    {
        const int mA = wave * 16 + (lane & 15);
#pragma unroll
        for (int c = 0; c < 2; ++c)
#pragma unroll
            for (int e = 0; e < 16; ++e)
                qf[c][e] = Qs[mA][c * 32 + a_kidx(e, hi)];
    }

    float mst[8], lst[8];
    v8f Oacc[4] = {};
#pragma unroll
    for (int r = 0; r < 8; ++r) { mst[r] = -3.0e38f; lst[r] = 0.0f; }

    for (int kt = 0; kt < Sn / 64; ++kt) {
        const int k0 = kt * 64;
        __syncthreads();   // all waves done reading Ks/Vs from prev iter
#pragma unroll
        for (int i = tid; i < 512; i += 128) {
            const int r = i >> 3, c8 = (i & 7) * 8;
            const unsigned loff = (unsigned)((r * PADA + c8) * 2);
            async_b128(ksOff + loff, Kp + (size_t)(k0 + r) * Dn + c8);
            async_b128(vsOff + loff, Vp + (size_t)(k0 + r) * Dn + c8);
        }
        if (tid < 64) kmb[tid] = kmask[(size_t)bb * Sn + k0 + tid] ? 0.0f : -1.0e9f;
        wait_async0();
        __syncthreads();

        // S = Q @ K^T * (1/181) + keymask_bias   (four 16x16 col tiles)
        v8f sc[4];
#pragma unroll
        for (int ct = 0; ct < 4; ++ct) {
            const int kcol = ct * 16 + (lane & 15);
            v16bf b0, b1;
#pragma unroll
            for (int e = 0; e < 16; ++e) {
                const int d0 = b_kidx(e, hi);
                b0[e] = Ks[kcol][d0];
                b1[e] = Ks[kcol][32 + d0];
            }
            v8f s = {};
            s = WMMA_BF16(qf[0], b0, s);
            s = WMMA_BF16(qf[1], b1, s);
            const float mb = kmb[kcol];
#pragma unroll
            for (int r = 0; r < 8; ++r) sc[ct][r] = s[r] * INV_SCALE + mb;
        }

        // online softmax row update (rows live across a 16-lane group)
#pragma unroll
        for (int r = 0; r < 8; ++r) {
            float v = fmaxf(fmaxf(sc[0][r], sc[1][r]), fmaxf(sc[2][r], sc[3][r]));
#pragma unroll
            for (int off = 1; off < 16; off <<= 1) v = fmaxf(v, __shfl_xor(v, off, 32));
            const float mnew = fmaxf(mst[r], v);
            const float alpha = __expf(mst[r] - mnew);
            float psum = 0.0f;
#pragma unroll
            for (int ct = 0; ct < 4; ++ct) {
                float p = __expf(sc[ct][r] - mnew);
                sc[ct][r] = p;
                psum += p;
            }
#pragma unroll
            for (int off = 1; off < 16; off <<= 1) psum += __shfl_xor(psum, off, 32);
            lst[r] = lst[r] * alpha + psum;
            mst[r] = mnew;
#pragma unroll
            for (int dt = 0; dt < 4; ++dt) Oacc[dt][r] *= alpha;
        }

        // P (C-layout f32) -> LDS bf16 -> A-layout fragments
#pragma unroll
        for (int r = 0; r < 8; ++r) {
            const int prow = r + (hi ? 8 : 0);
#pragma unroll
            for (int ct = 0; ct < 4; ++ct)
                Ps[wave][prow][ct * 16 + (lane & 15)] = (__bf16)sc[ct][r];
        }
        __syncthreads();

        v16bf ap[2];
#pragma unroll
        for (int c = 0; c < 2; ++c)
#pragma unroll
            for (int e = 0; e < 16; ++e)
                ap[c][e] = Ps[wave][lane & 15][c * 32 + a_kidx(e, hi)];

        // O += P @ V  (key dim 64 = two K32 chunks, four 16-wide d tiles)
#pragma unroll
        for (int dt = 0; dt < 4; ++dt) {
            const int dcol = dt * 16 + (lane & 15);
            v16bf bv0, bv1;
#pragma unroll
            for (int e = 0; e < 16; ++e) {
                const int kk2 = b_kidx(e, hi);
                bv0[e] = Vs[kk2][dcol];
                bv1[e] = Vs[32 + kk2][dcol];
            }
            Oacc[dt] = WMMA_BF16(ap[0], bv0, Oacc[dt]);
            Oacc[dt] = WMMA_BF16(ap[1], bv1, Oacc[dt]);
        }
    }

    // finalize: /rowsum, query-mask zero-fill, store ctx [B,S,H,D]
#pragma unroll
    for (int r = 0; r < 8; ++r) {
        const int qrow = q0 + wave * 16 + r + (hi ? 8 : 0);
        const float inv_l = 1.0f / lst[r];
        const float qm = qmask[(size_t)bb * Sn + qrow] ? 1.0f : 0.0f;
        const size_t obase = (((size_t)bb * Sn + qrow) * Hn + h) * Dn;
#pragma unroll
        for (int dt = 0; dt < 4; ++dt)
            Ctx[obase + dt * 16 + (lane & 15)] = __float2bfloat16(Oacc[dt][r] * inv_l * qm);
    }
}

// ---------------------------------------------------------------------------
// Kernel 3: out = (ctx @ Wo^T + bo) * query_mask, fp32 output.
// Same 32x64 tiling as kernel 1, ctx already bf16 (b128 copy to LDS).
// ---------------------------------------------------------------------------
__global__ __launch_bounds__(256)
void proj_out_kernel(const __hip_bfloat16* __restrict__ CtxG, const float* __restrict__ W,
                     const float* __restrict__ bias, const unsigned char* __restrict__ rmask,
                     float* __restrict__ Out) {
    using namespace mha;
    __shared__ __align__(16) __bf16 As[32][PADP];
    __shared__ __align__(16) __bf16 Bs[64][PADP];

    const int tid  = threadIdx.x;
    const int wave = tid >> 5;
    const int lane = tid & 31;
    const bool hi  = lane >= 16;
    const int tm = wave >> 2, tn = wave & 3;
    const int m0 = blockIdx.x * 32;
    const int n0 = blockIdx.y * 64;
    const __bf16* Ctx = reinterpret_cast<const __bf16*>(CtxG);

    v8f acc = {};
    for (int kk = 0; kk < En; kk += 32) {
        if (tid < 128) {   // A: 32x32 bf16, 128 x 16B chunks
            const int r = tid >> 2, c8 = (tid & 3) * 8;
            *(float4*)&As[r][c8] =
                *(const float4*)&Ctx[(size_t)(m0 + r) * En + kk + c8];
        }
#pragma unroll
        for (int i = tid; i < 512; i += 256) {  // B: 64x32 fp32 -> bf16
            const int r = i >> 3, c4 = (i & 7) * 4;
            const float4 wv = *(const float4*)&W[(size_t)(n0 + r) * En + kk + c4];
            cvt_store4(&Bs[r][c4], wv);
        }
        __syncthreads();
        v16bf a, b;
        const int mA = tm * 16 + (lane & 15);
        const int nB = tn * 16 + (lane & 15);
#pragma unroll
        for (int e = 0; e < 16; ++e) {
            a[e] = As[mA][a_kidx(e, hi)];
            b[e] = Bs[nB][b_kidx(e, hi)];
        }
        acc = WMMA_BF16(a, b, acc);
        __syncthreads();
    }

    const int n = n0 + tn * 16 + (lane & 15);
    const float bv = bias[n];
#pragma unroll
    for (int r = 0; r < 8; ++r) {
        const int gm = m0 + tm * 16 + r + (hi ? 8 : 0);
        const float mk = rmask[gm] ? 1.0f : 0.0f;
        Out[(size_t)gm * En + n] = (acc[r] + bv) * mk;
    }
}

// ---------------------------------------------------------------------------
extern "C" void kernel_launch(void* const* d_in, const int* in_sizes, int n_in,
                              void* d_out, int out_size, void* d_ws, size_t ws_size,
                              hipStream_t stream) {
    using namespace mha;
    const float* q  = (const float*)d_in[0];
    const float* k  = (const float*)d_in[1];
    const float* v  = (const float*)d_in[2];
    const float* Wq = (const float*)d_in[3];
    const float* bq = (const float*)d_in[4];
    const float* Wk = (const float*)d_in[5];
    const float* bk = (const float*)d_in[6];
    const float* Wv = (const float*)d_in[7];
    const float* bv = (const float*)d_in[8];
    const float* Wo = (const float*)d_in[9];
    const float* bo = (const float*)d_in[10];
    const float* gq = (const float*)d_in[11];
    const float* betaq = (const float*)d_in[12];
    const float* gk = (const float*)d_in[13];
    const float* betak = (const float*)d_in[14];
    const float* gv = (const float*)d_in[15];
    const float* betav = (const float*)d_in[16];
    const unsigned char* query_mask = (const unsigned char*)d_in[17];
    const unsigned char* key_mask   = (const unsigned char*)d_in[18];
    const unsigned char* value_mask = (const unsigned char*)d_in[19];

    const size_t headBytes = (size_t)Bn * Hn * Sn * Dn * sizeof(__hip_bfloat16); // 8 MB
    char* ws = (char*)d_ws;
    __hip_bfloat16* Qh  = (__hip_bfloat16*)(ws);
    __hip_bfloat16* Kh  = (__hip_bfloat16*)(ws + headBytes);
    __hip_bfloat16* Vh  = (__hip_bfloat16*)(ws + 2 * headBytes);
    __hip_bfloat16* Ctx = (__hip_bfloat16*)(ws + 3 * headBytes);

    const dim3 gProj((Bn * Sn) / 32, Hn);   // 256 x 8
    proj_ln_kernel<<<gProj, 256, 0, stream>>>(q, Wq, bq, query_mask, gq, betaq, Qh);
    proj_ln_kernel<<<gProj, 256, 0, stream>>>(k, Wk, bk, key_mask,   gk, betak, Kh);
    proj_ln_kernel<<<gProj, 256, 0, stream>>>(v, Wv, bv, value_mask, gv, betav, Vh);

    const dim3 gAttn(Sn / 64, Bn * Hn);     // 32 x 32
    attn_kernel<<<gAttn, 128, 0, stream>>>(Qh, Kh, Vh, query_mask, key_mask, Ctx);

    const dim3 gOut((Bn * Sn) / 32, En / 64);  // 256 x 8
    proj_out_kernel<<<gOut, 256, 0, stream>>>(Ctx, Wo, bo, query_mask, (float*)d_out);
}
